// NodeDetector_rnn_77979426226962
// MI455X (gfx1250) — matrix operand order repaired and checked by reading, not compiled, not run on hardware
//
#include <hip/hip_runtime.h>
#include <math.h>

typedef __attribute__((ext_vector_type(16))) _Float16 v16h;
typedef __attribute__((ext_vector_type(8)))  _Float16 v8h;
typedef __attribute__((ext_vector_type(4)))  _Float16 v4h;
typedef __attribute__((ext_vector_type(8)))  float    v8f;
typedef __attribute__((ext_vector_type(4)))  float    v4f;

#define DEV __device__ __forceinline__

// ---- WMMA fragment loaders (ISA 7.12.2 layouts, wave32) ----
DEV v16h frag_a(const _Float16* A, int lda, int row0, int k0, int lane) {
  int m = row0 + (lane & 15);
  int koff = (lane < 16) ? 0 : 8;
  const _Float16* p = A + m * lda + k0 + koff;
  v8h lo = *(const v8h*)(p);
  v8h hi = *(const v8h*)(p + 16);
  return __builtin_shufflevector(lo, hi, 0,1,2,3,4,5,6,7,8,9,10,11,12,13,14,15);
}
DEV v16h frag_b(const _Float16* BT, int ldb, int col0, int k0, int lane) {
  int n = col0 + (lane & 15);
  int koff = (lane < 16) ? 0 : 8;
  const _Float16* p = BT + n * ldb + k0 + koff;
  v8h lo = *(const v8h*)(p);
  v8h hi = *(const v8h*)(p + 16);
  return __builtin_shufflevector(lo, hi, 0,1,2,3,4,5,6,7,8,9,10,11,12,13,14,15);
}

// Cooperative LDS GEMM: C[M][N] f32 = A[M][K] f16 * BT[N][K] f16, 8 waves / 256 thr.
DEV void lds_gemm(float* C, int ldc, const _Float16* A, int lda,
                  const _Float16* BT, int ldb, int M, int N, int K) {
  int wave = threadIdx.x >> 5, lane = threadIdx.x & 31;
  int tn_ = N >> 4, tm_ = M >> 4, nt = tm_ * tn_;
  for (int tile = wave; tile < nt; tile += 8) {
    int tm = tile / tn_, tn = tile - tm * tn_;
    v8f acc = {0.f,0.f,0.f,0.f,0.f,0.f,0.f,0.f};
    for (int k0 = 0; k0 < K; k0 += 32) {
      v16h a = frag_a(A, lda, tm * 16, k0, lane);
      v16h b = frag_b(BT, ldb, tn * 16, k0, lane);
      acc = __builtin_amdgcn_wmma_f32_16x16x32_f16(false, a, false, b,
                                                   (short)0, acc, false, false);
    }
    int col = tn * 16 + (lane & 15);
    int rb  = tm * 16 + ((lane >> 4) << 3);
#pragma unroll
    for (int r = 0; r < 8; ++r) C[(rb + r) * ldc + col] = acc[r];
  }
}

// Vectorized f32 -> f16 staging: 8 contiguous halfs / thread-iter -> ds_store_b128.
// n must be a multiple of 2048.
DEV void stage16(_Float16* dst, const float* src, int n, int tid) {
  for (int i = tid * 8; i < n; i += 256 * 8) {
    v4f a = *(const v4f*)(src + i);
    v4f b = *(const v4f*)(src + i + 4);
    v4h ha = __builtin_convertvector(a, v4h);
    v4h hb = __builtin_convertvector(b, v4h);
    *(v8h*)(dst + i) = __builtin_shufflevector(ha, hb, 0,1,2,3,4,5,6,7);
  }
}
DEV void lds16(_Float16* dst, const float* srcLds, int n, int tid) {
  for (int i = tid * 8; i < n; i += 256 * 8) {
    v4f a = *(const v4f*)(srcLds + i);
    v4f b = *(const v4f*)(srcLds + i + 4);
    v4h ha = __builtin_convertvector(a, v4h);
    v4h hb = __builtin_convertvector(b, v4h);
    *(v8h*)(dst + i) = __builtin_shufflevector(ha, hb, 0,1,2,3,4,5,6,7);
  }
}
// Transposed staging (scattered by nature; only used for 4 small matrices).
DEV void stage16T(_Float16* dst, const float* src, int rows, int cols, int tid) {
  for (int i = tid; i < rows * cols; i += 256) {
    int r = i / cols, c = i - r * cols;
    dst[c * rows + r] = (_Float16)src[i];
  }
}

// =====================================================================
// Kernel 1: per step t (8 blocks): precompute XL/XR off/diag rows.
// =====================================================================
__global__ __launch_bounds__(256) void k1_precompute(
    const float* __restrict__ x, const float* __restrict__ E,
    const float* __restrict__ nodep, const float* __restrict__ embp,
    const float* __restrict__ w0, const float* __restrict__ w1,
    const float* __restrict__ b1, const float* __restrict__ l2w,
    const float* __restrict__ l2b, const float* __restrict__ mp,
    const float* __restrict__ np, const float* __restrict__ wl1,
    const float* __restrict__ wr1,
    float* __restrict__ XLo, float* __restrict__ XLd,
    float* __restrict__ XRo, float* __restrict__ XRd) {
  __shared__ __align__(16) char smem[128 * 1024];
  _Float16* A16 = (_Float16*)(smem);              // 64x128 f16
  _Float16* B16 = (_Float16*)(smem + 16 * 1024);  // 128x128 f16 (BT layout)
  float*    C32 = (float*)(smem + 48 * 1024);     // 64x128 f32
  float*    D32 = (float*)(smem + 80 * 1024);     // 64x128 f32
  _Float16* E16 = (_Float16*)(smem + 112 * 1024); // 64x128 f16 (diag path)
  const int t = blockIdx.x, tid = threadIdx.x;

  // pull hot weights into GL2 early (global_prefetch_b8)
  __builtin_prefetch(w0 + tid * 64, 0, 1);
  __builtin_prefetch(w1 + tid * 64, 0, 1);

  // 1) Ep = E[t] @ embp  -> C32 [64][128]
  stage16(A16, E + t * 64 * 64, 64 * 64, tid);
  stage16T(B16, embp, 64, 128, tid);
  __syncthreads();
  lds_gemm(C32, 128, A16, 64, B16, 64, 64, 128, 64);
  __syncthreads();
  // 2) A1 = Ep @ w0^T -> D32
  lds16(A16, C32, 64 * 128, tid);
  stage16(B16, w0, 128 * 128, tid);
  __syncthreads();
  lds_gemm(D32, 128, A16, 128, B16, 128, 64, 128, 128);
  __syncthreads();
  // 3) xp = x[t+1] @ nodep -> C32
  stage16(A16, x + (t + 1) * 64 * 64, 64 * 64, tid);
  stage16T(B16, nodep, 64, 128, tid);
  __syncthreads();
  lds_gemm(C32, 128, A16, 64, B16, 64, 64, 128, 64);
  __syncthreads();
  // 4) A2 = xp @ w1^T -> C32
  lds16(A16, C32, 64 * 128, tid);
  stage16(B16, w1, 128 * 128, tid);
  __syncthreads();
  lds_gemm(C32, 128, A16, 128, B16, 128, 64, 128, 128);
  __syncthreads();
  // 5) h_full -> A16, h_diag -> E16 (vectorized: 8 contiguous elems/thread)
  for (int i0 = tid * 8; i0 < 64 * 128; i0 += 256 * 8) {
    float vf[8], vd[8];
#pragma unroll
    for (int j = 0; j < 8; ++j) {
      int i = i0 + j, c = i & 127;
      float a1 = D32[i] + b1[c];
      vf[j] = tanhf(a1 + C32[i]);
      vd[j] = tanhf(a1);
    }
    v8h hf, hd;
#pragma unroll
    for (int j = 0; j < 8; ++j) { hf[j] = (_Float16)vf[j]; hd[j] = (_Float16)vd[j]; }
    *(v8h*)(A16 + i0) = hf;
    *(v8h*)(E16 + i0) = hd;
  }
  __syncthreads();
  // 6) h2 = h @ l2w^T  (K=128 -> 64)
  stage16(B16, l2w, 64 * 128, tid);
  __syncthreads();
  lds_gemm(C32, 64, A16, 128, B16, 128, 64, 64, 128);
  lds_gemm(D32, 64, E16, 128, B16, 128, 64, 64, 128);
  __syncthreads();
  // 7) + lin2_b, convert to f16
  for (int i0 = tid * 8; i0 < 64 * 64; i0 += 256 * 8) {
    v8h hf, hd;
#pragma unroll
    for (int j = 0; j < 8; ++j) {
      int i = i0 + j, d = i & 63;
      hf[j] = (_Float16)(C32[i] + l2b[d]);
      hd[j] = (_Float16)(D32[i] + l2b[d]);
    }
    *(v8h*)(A16 + i0) = hf;
    *(v8h*)(E16 + i0) = hd;
  }
  __syncthreads();
  // 8) P_off = h2_full @ nproj -> C32 ; P_diag = h2_diag @ mproj -> D32
  stage16T(B16, np, 64, 64, tid);
  __syncthreads();
  lds_gemm(C32, 64, A16, 64, B16, 64, 64, 64, 64);
  __syncthreads();
  stage16T(B16, mp, 64, 64, tid);
  __syncthreads();
  lds_gemm(D32, 64, E16, 64, B16, 64, 64, 64, 64);
  __syncthreads();
  // 9) P -> f16
  lds16(A16, C32, 64 * 64, tid);
  lds16(E16, D32, 64 * 64, tid);
  __syncthreads();
  // 10) head GEMMs -> global workspace
  for (int h = 0; h < 4; ++h) {
    for (int wsel = 0; wsel < 2; ++wsel) {
      const float* W  = (wsel == 0 ? wl1 : wr1) + h * 64 * 64;
      float* Go = (wsel == 0 ? XLo : XRo) + (t * 4 + h) * 64 * 64;
      float* Gd = (wsel == 0 ? XLd : XRd) + (t * 4 + h) * 64 * 64;
      stage16(B16, W, 64 * 64, tid);
      __syncthreads();
      lds_gemm(C32, 64, A16, 64, B16, 64, 64, 64, 64);
      lds_gemm(D32, 64, E16, 64, B16, 64, 64, 64, 64);
      __syncthreads();
      for (int i = tid * 4; i < 64 * 64; i += 256 * 4) {
        *(v4f*)(Go + i) = *(const v4f*)(C32 + i);
        *(v4f*)(Gd + i) = *(const v4f*)(D32 + i);
      }
      __syncthreads();
    }
  }
}

// =====================================================================
// Kernel 2: one block per (step t, graph b). 512 blocks, 256 threads.
// =====================================================================
__global__ __launch_bounds__(256) void k2_graph(
    const int* __restrict__ edge,
    const float* __restrict__ XLo, const float* __restrict__ XLd,
    const float* __restrict__ XRo, const float* __restrict__ XRd,
    const float* __restrict__ att1, const float* __restrict__ b1g,
    const float* __restrict__ wl2, const float* __restrict__ wr2,
    const float* __restrict__ att2, const float* __restrict__ b2g,
    const float* __restrict__ reconw, const float* __restrict__ reconb,
    float* __restrict__ out) {
  __shared__ __align__(16) char smem[118 * 1024];
  float*         sXL  = (float*)(smem);                    // [4][64][64] 64KB
  unsigned char* sM   = (unsigned char*)(smem + 65536);    // [64][64] mask 4KB
  float*         sA   = (float*)(smem + 69632);            // [4][64] att (1 then 2)
  _Float16*      sG1h = (_Float16*)(smem + 70656);         // [64][72] g1 f16
  _Float16*      sW2h = (_Float16*)(smem + 79872);         // [4][64][64] wl2 f16 32KB
  float*         sXr2 = (float*)(smem + 112640);           // [4][64]
  float*         sLg2 = (float*)(smem + 113664);           // [4][64]
  float*         sO2  = (float*)(smem + 114688);           // [4][64]
  float*         sG2  = (float*)(smem + 115712);           // [64]
  const int tid = threadIdx.x;
  const int t = blockIdx.x >> 6, b = blockIdx.x & 63;
  const int tn = tid & 63;   // node index (role varies by phase)
  const int hh = tid >> 6;   // head

  __builtin_prefetch(wl2 + tid * 16, 0, 1);
  __builtin_prefetch(wr2 + tid * 16, 0, 1);

  // ---- stage xl1 rows (row b swapped to diag variant), mask, att1, wl2 f16
  {
    const float* src = (tn == b) ? (XLd + ((t * 4 + hh) * 64 + b) * 64)
                                 : (XLo + ((t * 4 + hh) * 64 + tn) * 64);
    float* dst = sXL + (hh * 64 + tn) * 64;
    for (int e = 0; e < 64; e += 4) *(v4f*)(dst + e) = *(const v4f*)(src + e);
  }
  for (int i = tid; i < 4096; i += 256) {
    int s_ = i >> 6, d_ = i & 63;
    sM[i] = (edge[((t + 1) * 64 + s_) * 64 + d_] != 0) || (s_ == d_);
  }
  sA[tid] = att1[tid];
  stage16(sW2h, wl2, 16384, tid);
  // xr row for this (target=tn, head=hh) into registers
  float xr[64];
  {
    const float* src = (tn == b) ? (XRd + ((t * 4 + hh) * 64 + b) * 64)
                                 : (XRo + ((t * 4 + hh) * 64 + tn) * 64);
#pragma unroll
    for (int e = 0; e < 64; ++e) xr[e] = src[e];
  }
  __syncthreads();

  // ---- gat1 attention: online softmax over sources s, thread = (target tn, head hh)
  const float* xlh = sXL + hh * 4096;
  const float* ath = sA + hh * 64;
  float mrun = -1e30f, drun = 0.f;
  for (int s = 0; s < 64; ++s) {
    if (!sM[s * 64 + tn]) continue;
    float acc = 0.f;
    const float* xls = xlh + s * 64;
#pragma unroll
    for (int e = 0; e < 64; ++e) {
      float z = xls[e] + xr[e];
      float l = fmaxf(z, 0.f) + 0.2f * fminf(z, 0.f);  // leaky_relu(0.2)
      acc = fmaf(ath[e], l, acc);
    }
    float mn = fmaxf(mrun, acc);
    drun = drun * __expf(mrun - mn) + __expf(acc - mn);
    mrun = mn;
  }
  float inv = 1.f / drun;
  float og[64];
#pragma unroll
  for (int e = 0; e < 64; ++e) og[e] = 0.f;
  for (int s = 0; s < 64; ++s) {
    if (!sM[s * 64 + tn]) continue;
    float acc = 0.f;
    const float* xls = xlh + s * 64;
#pragma unroll
    for (int e = 0; e < 64; ++e) {
      float z = xls[e] + xr[e];
      float l = fmaxf(z, 0.f) + 0.2f * fminf(z, 0.f);
      acc = fmaf(ath[e], l, acc);
    }
    float a = __expf(acc - mrun) * inv;
#pragma unroll
    for (int e = 0; e < 64; ++e) og[e] = fmaf(a, xls[e], og[e]);
  }
  __syncthreads();
  // stage per-head outputs back into sXL [h][t][e] (vector stores)
  {
    float* dst = sXL + (hh * 64 + tn) * 64;
#pragma unroll
    for (int e = 0; e < 64; e += 4) {
      v4f v; v[0] = og[e]; v[1] = og[e + 1]; v[2] = og[e + 2]; v[3] = og[e + 3];
      *(v4f*)(dst + e) = v;
    }
  }
  __syncthreads();
  // g1 = elu(mean_h + b1g) -> f16, packed b128 stores (thread: 16 contig elems)
  {
    int n_ = tid >> 2, e0 = (tid & 3) * 16;
    float vb[16];
#pragma unroll
    for (int j = 0; j < 16; ++j) {
      int e_ = e0 + j;
      float v = 0.25f * (sXL[(0 * 64 + n_) * 64 + e_] + sXL[(1 * 64 + n_) * 64 + e_] +
                         sXL[(2 * 64 + n_) * 64 + e_] + sXL[(3 * 64 + n_) * 64 + e_]) + b1g[e_];
      vb[j] = (v > 0.f) ? v : (__expf(v) - 1.f);
    }
    v8h h0, h1;
#pragma unroll
    for (int j = 0; j < 8; ++j) { h0[j] = (_Float16)vb[j]; h1[j] = (_Float16)vb[8 + j]; }
    *(v8h*)(sG1h + n_ * 72 + e0) = h0;
    *(v8h*)(sG1h + n_ * 72 + e0 + 8) = h1;
  }
  __syncthreads();

  // ---- gat2 source features: xl2[h] = g1 @ wl2[h]^T via WMMA (overwrites sXL)
  for (int h_ = 0; h_ < 4; ++h_)
    lds_gemm(sXL + h_ * 4096, 64, sG1h, 72, sW2h + h_ * 4096, 64, 64, 64, 64);
  __syncthreads();

  // swap in att2; xr2 for target row b only
  sA[tid] = att2[tid];
  {
    float acc = 0.f;
    const float* W = wr2 + (hh * 64 + tn) * 64;
#pragma unroll
    for (int d_ = 0; d_ < 64; ++d_) acc = fmaf(W[d_], (float)sG1h[b * 72 + d_], acc);
    sXr2[hh * 64 + tn] = acc;
  }
  __syncthreads();
  // logits2[s=tn, h=hh] for target b
  {
    float lg = -1e30f;
    if (sM[tn * 64 + b]) {
      float acc = 0.f;
      const float* xls = sXL + (hh * 64 + tn) * 64;
      const float* xrb = sXr2 + hh * 64;
      const float* at2 = sA + hh * 64;
#pragma unroll
      for (int e = 0; e < 64; ++e) {
        float z = xls[e] + xrb[e];
        float l = fmaxf(z, 0.f) + 0.2f * fminf(z, 0.f);
        acc = fmaf(at2[e], l, acc);
      }
      lg = acc;
    }
    sLg2[hh * 64 + tn] = lg;
  }
  __syncthreads();
  if (tid < 4) {  // softmax over s per head
    float m = -1e30f;
    for (int s = 0; s < 64; ++s) m = fmaxf(m, sLg2[tid * 64 + s]);
    float d_ = 0.f;
    for (int s = 0; s < 64; ++s) d_ += __expf(sLg2[tid * 64 + s] - m);
    float i2 = 1.f / d_;
    for (int s = 0; s < 64; ++s) sLg2[tid * 64 + s] = __expf(sLg2[tid * 64 + s] - m) * i2;
  }
  __syncthreads();
  {  // out2[h][e=tn] = sum_s alpha2[s,h] * xl2[h][s][e]
    float acc = 0.f;
    for (int s = 0; s < 64; ++s)
      acc = fmaf(sLg2[hh * 64 + s], sXL[(hh * 64 + s) * 64 + tn], acc);
    sO2[hh * 64 + tn] = acc;
  }
  __syncthreads();
  if (tid < 64) {  // g2[b] row = elu(mean_h + b2g)
    float v = 0.25f * (sO2[tid] + sO2[64 + tid] + sO2[128 + tid] + sO2[192 + tid]) + b2g[tid];
    v = (v > 0.f) ? v : (__expf(v) - 1.f);
    sG2[tid] = v;
  }
  __syncthreads();
  if (tid < 64) {  // recon + tanh
    float acc = reconb[tid];
    const float* W = reconw + tid * 64;
#pragma unroll
    for (int e = 0; e < 64; ++e) acc = fmaf(W[e], sG2[e], acc);
    out[(t * 64 + b) * 64 + tid] = tanhf(acc);
  }
}

extern "C" void kernel_launch(void* const* d_in, const int* in_sizes, int n_in,
                              void* d_out, int out_size, void* d_ws, size_t ws_size,
                              hipStream_t stream) {
  (void)in_sizes; (void)n_in; (void)out_size; (void)ws_size;
  const float* x     = (const float*)d_in[0];
  const int*   edge  = (const int*)d_in[1];
  const float* E     = (const float*)d_in[2];
  const float* nodep = (const float*)d_in[3];
  const float* embp  = (const float*)d_in[4];
  const float* w0    = (const float*)d_in[5];
  const float* w1    = (const float*)d_in[6];
  const float* b1    = (const float*)d_in[7];
  const float* l2w   = (const float*)d_in[8];
  const float* l2b   = (const float*)d_in[9];
  const float* mp    = (const float*)d_in[10];
  const float* np    = (const float*)d_in[11];
  const float* wl1   = (const float*)d_in[12];
  const float* wr1   = (const float*)d_in[13];
  const float* a1    = (const float*)d_in[14];
  const float* g1b   = (const float*)d_in[15];
  const float* wl2   = (const float*)d_in[16];
  const float* wr2   = (const float*)d_in[17];
  const float* a2    = (const float*)d_in[18];
  const float* g2b   = (const float*)d_in[19];
  const float* rw    = (const float*)d_in[20];
  const float* rb    = (const float*)d_in[21];

  float* ws = (float*)d_ws;
  const size_t blk = (size_t)8 * 4 * 64 * 64;
  float* XLo = ws;
  float* XLd = ws + blk;
  float* XRo = ws + 2 * blk;
  float* XRd = ws + 3 * blk;

  k1_precompute<<<8, 256, 0, stream>>>(x, E, nodep, embp, w0, w1, b1, l2w, l2b,
                                       mp, np, wl1, wr1, XLo, XLd, XRo, XRd);
  k2_graph<<<512, 256, 0, stream>>>(edge, XLo, XLd, XRo, XRd, a1, g1b,
                                    wl2, wr2, a2, g2b, rw, rb, (float*)d_out);
}
